// MMOELoraLinear_78288663872094
// MI455X (gfx1250) — compile-verified, matrix operand order checked by reading, not run on hardware
//
#include <hip/hip_runtime.h>

// ============================================================================
// MMOELoraLinear for MI455X (gfx1250, wave32, WMMA)
//
// out = x @ W^T + b + SCALING * sum_e gw[b,e] * (x @ Wa_e^T) @ Wb_e^T
//   M = 16384, N = 4096, K = 4096, total rank 64 (8 experts x 8)
//
// Base GEMM = 550 GFLOP (compute-bound; AI > 1000 FLOP/B at 23.3 TB/s HBM)
//   -> v_wmma_f32_16x16x32_bf16, fp32 accumulate.
// LoRA (3% of FLOPs) is folded into the SAME accumulators as 2 extra K-steps:
//   hg[m, e*8+r] = SCALING * gw[batch(m), e] * (x @ Wa^T)[m, e*8+r]   (bf16)
//   delta        = hg @ Wb_all^T,  Wb_all[o, e*8+r]  (batch-independent)
//
// CDNA5 data path: LDS staging uses GLOBAL_LOAD_ASYNC_TO_LDS_B128 (ASYNCcnt,
// no VGPR round-trip) + s_wait_asynccnt, with global_prefetch_b8 4 steps ahead.
//
// Workspace layout (bytes), ~171 MB total:
//   xb     @ 0          : bf16 x        [16384,4096]  (134217728)
//   wb16   @ 134217728  : bf16 base_W   [4096,4096]   ( 33554432)
//   wab    @ 167772160  : bf16 Wa       [64,4096]     (   524288)
//   wball  @ 168296448  : bf16 Wb_all   [4096,64]     (   524288)
//   hg     @ 168820736  : bf16 hg       [16384,64]    (  2097152)
//   gw     @ 170917888  : f32  gate     [8,8]         (      256)
// ============================================================================

#define BB   8
#define SS   2048
#define DIN  4096
#define DOUT 4096
#define MM   (BB * SS)
#define NE   8
#define RE   8
#define RTOT 64
#define SCALING (16.0f / 64.0f)

typedef __attribute__((ext_vector_type(16))) __bf16          v16bf;
typedef __attribute__((ext_vector_type(8)))  float           v8f;
typedef __attribute__((ext_vector_type(8)))  unsigned short  us8;
typedef __attribute__((ext_vector_type(4)))  unsigned short  us4;

__device__ __forceinline__ unsigned short f2bf(float f) {
  union { float f; unsigned int u; } v; v.f = f;
  unsigned int u = v.u;
  return (unsigned short)((u + 0x7FFFu + ((u >> 16) & 1u)) >> 16);  // RNE
}

// 16-bit A/B fragment for v_wmma_f32_16x16x32_bf16 (ISA 7.12.2 layout):
// lane<16: K = {kh..kh+7, kh+16..kh+23} with kh=0 ; lane>=16: kh=8.
// Two 16-byte loads per lane (global_load_b128 / ds_load_b128).
__device__ __forceinline__ v16bf load_frag(const unsigned short* p) {
  union { v16bf v; us8 h[2]; } u;
  u.h[0] = *(const us8*)(p);
  u.h[1] = *(const us8*)(p + 16);
  return u.v;
}

// ---------------------------------------------------------------------------
// fp32 -> bf16 bulk convert (vectorized x4)
// ---------------------------------------------------------------------------
__global__ void cvt_f32_bf16_kernel(const float* __restrict__ in,
                                    unsigned short* __restrict__ out, int n4) {
  int i = blockIdx.x * blockDim.x + threadIdx.x;
  if (i >= n4) return;
  float4 f = ((const float4*)in)[i];
  us4 o;
  o.x = f2bf(f.x); o.y = f2bf(f.y); o.z = f2bf(f.z); o.w = f2bf(f.w);
  ((us4*)out)[i] = o;
}

// ---------------------------------------------------------------------------
// Wb [E, DOUT, RE] -> Wb_all bf16 [DOUT, E*RE]
// ---------------------------------------------------------------------------
__global__ void wb_permute_kernel(const float* __restrict__ Wb,
                                  unsigned short* __restrict__ wball) {
  int i = blockIdx.x * blockDim.x + threadIdx.x;     // over E*DOUT*RE = 262144
  if (i >= NE * DOUT * RE) return;
  int r = i & (RE - 1);
  int o = (i >> 3) & (DOUT - 1);
  int e = i >> 15;                                   // 4096*8 = 2^15
  wball[o * RTOT + e * RE + r] = f2bf(Wb[i]);
}

// ---------------------------------------------------------------------------
// gate: gw[b,e] = softmax_e( dot(task_emb[task_id[b]], gate_W[e]) )
// ---------------------------------------------------------------------------
__global__ void gate_kernel(const int* __restrict__ task_id,
                            const float* __restrict__ task_emb,
                            const float* __restrict__ gate_W,
                            float* __restrict__ gw) {
  __shared__ float logits[BB][NE];
  int t = threadIdx.x;                 // 64 threads: b = t>>3, e = t&7
  int b = t >> 3, e = t & 7;
  const float* te = task_emb + task_id[b] * 64;
  const float* g  = gate_W + e * 64;
  float acc = 0.f;
  #pragma unroll 8
  for (int k = 0; k < 64; ++k) acc += te[k] * g[k];
  logits[b][e] = acc;
  __syncthreads();
  float m = logits[b][0];
  #pragma unroll
  for (int j = 1; j < NE; ++j) m = fmaxf(m, logits[b][j]);
  float s = 0.f;
  #pragma unroll
  for (int j = 0; j < NE; ++j) s += __expf(logits[b][j] - m);
  gw[t] = __expf(acc - m) / s;
}

// ---------------------------------------------------------------------------
// hg[m, er] = SCALING * gw[batch(m), er/8] * sum_k xb[m,k] * wab[er,k]
// One wave per 16-row tile, all 64 columns (4 WMMA tiles), K = 4096.
// ---------------------------------------------------------------------------
__global__ __launch_bounds__(128) void lora_h_kernel(
    const unsigned short* __restrict__ xb,   // [M, DIN] bf16
    const unsigned short* __restrict__ wab,  // [64, DIN] bf16
    const float* __restrict__ gw,            // [8,8]
    unsigned short* __restrict__ hg) {       // [M, 64] bf16
  int lane  = threadIdx.x & 31;
  int wave  = threadIdx.x >> 5;
  int mtile = blockIdx.x * 4 + wave;         // 0..1023
  int lh    = lane & 15;
  int kh    = (lane < 16) ? 0 : 8;

  const unsigned short* arow = xb + (size_t)(mtile * 16 + lh) * DIN + kh;
  const unsigned short* brow = wab + (size_t)lh * DIN + kh;

  v8f acc[4];
  #pragma unroll
  for (int nt = 0; nt < 4; ++nt) acc[nt] = (v8f)(0.0f);

  for (int k0 = 0; k0 < DIN; k0 += 32) {
    v16bf a = load_frag(arow + k0);
    #pragma unroll
    for (int nt = 0; nt < 4; ++nt) {
      v16bf bfrag = load_frag(brow + (size_t)(nt * 16) * DIN + k0);
      acc[nt] = __builtin_amdgcn_wmma_f32_16x16x32_bf16(
          false, a, false, bfrag, (short)0, acc[nt], false, false);
    }
  }

  int batch = (mtile * 16) >> 11;            // row / 2048 (tile is batch-pure)
  int rbase = mtile * 16 + ((lane >> 4) << 3);
  #pragma unroll
  for (int nt = 0; nt < 4; ++nt) {
    int col = nt * 16 + lh;
    float f = SCALING * gw[batch * NE + (col >> 3)];
    #pragma unroll
    for (int i = 0; i < 8; ++i)
      hg[(size_t)(rbase + i) * RTOT + col] = f2bf(acc[nt][i] * f);
  }
}

// ---------------------------------------------------------------------------
// Main fused GEMM: out = xb @ wb16^T (+ hg @ wball^T as 2 extra K-steps) + b
// Block 128x128, 8 waves, wave = 32x64 (2x4 accum tiles), double-buffered LDS.
// Staging: GLOBAL_LOAD_ASYNC_TO_LDS_B128 (GVS mode: SGPR base + VGPR offset),
// synchronized with s_wait_asynccnt + workgroup barrier.
// LDS row stride 56 halves = 112 B: 16B-aligned, 28-bank stride -> the 16 rows
// of a fragment read hit 16 distinct bank groups.
// ---------------------------------------------------------------------------
#define BM  128
#define BN  128
#define LDT 56

__global__ __launch_bounds__(256) void mmoe_gemm_kernel(
    const unsigned short* __restrict__ xb,     // [M, DIN]
    const unsigned short* __restrict__ wb16,   // [DOUT, DIN]
    const unsigned short* __restrict__ hg,     // [M, 64]
    const unsigned short* __restrict__ wball,  // [DOUT, 64]
    const float* __restrict__ bias,            // [DOUT]
    float* __restrict__ out) {                 // [M, DOUT]
  __shared__ __attribute__((aligned(16))) unsigned short As[2][BM][LDT];
  __shared__ __attribute__((aligned(16))) unsigned short Bs[2][BN][LDT];

  int tid   = threadIdx.x;
  int lane  = tid & 31;
  int wave  = tid >> 5;
  int waveM = wave & 3;                        // 4 waves along M (4*32 = 128)
  int waveN = wave >> 2;                       // 2 waves along N (2*64 = 128)
  int blockM = blockIdx.y * BM;
  int blockN = blockIdx.x * BN;
  int lh = lane & 15;
  int kh = (lane < 16) ? 0 : 8;

  const unsigned short* aG = xb    + (size_t)blockM * DIN;
  const unsigned short* bG = wb16  + (size_t)blockN * DIN;
  const unsigned short* aL = hg    + (size_t)blockM * RTOT;
  const unsigned short* bL = wball + (size_t)blockN * RTOT;

  v8f acc[2][4];
  #pragma unroll
  for (int mt = 0; mt < 2; ++mt)
    #pragma unroll
    for (int nt = 0; nt < 4; ++nt) acc[mt][nt] = (v8f)(0.0f);

  // 128 rows x 32 halves tile = 512 chunks of 16 B; 2 async copies per thread.
  // GLOBAL_LOAD_ASYNC_TO_LDS_B128, GVS: mem = SGPR64 + VGPR32 ; LDS dest VGPR.
  auto async_tile = [&](const unsigned short* src, int ld,
                        unsigned short (*dst)[LDT]) {
    #pragma unroll
    for (int c = tid; c < 512; c += 256) {
      int r  = c >> 2;
      int cc = (c & 3) << 3;
      unsigned ldsAddr = (unsigned)(size_t)&dst[r][cc];
      unsigned gOff    = (unsigned)(((unsigned)r * (unsigned)ld + (unsigned)cc) * 2u);
      asm volatile("global_load_async_to_lds_b128 %0, %1, %2"
                   :
                   : "v"(ldsAddr), "v"(gOff), "s"(src)
                   : "memory");
    }
  };

  async_tile(aG, DIN, As[0]);
  async_tile(bG, DIN, Bs[0]);
  asm volatile("s_wait_asynccnt 0x0" ::: "memory");
  __syncthreads();

  int buf = 0;
  const int KSTEPS = DIN / 32;                 // 128 base steps + 2 LoRA steps
  for (int ks = 0; ks < KSTEPS + 2; ++ks) {
    int nbuf = buf ^ 1;
    if (ks + 1 < KSTEPS) {                     // async-prefetch next base tile
      async_tile(aG + (ks + 1) * 32, DIN, As[nbuf]);
      async_tile(bG + (ks + 1) * 32, DIN, Bs[nbuf]);
    } else if (ks + 1 < KSTEPS + 2) {          // async-prefetch LoRA tiles
      int lk = (ks + 1 - KSTEPS) * 32;
      async_tile(aL + lk, RTOT, As[nbuf]);
      async_tile(bL + lk, RTOT, Bs[nbuf]);
    }
    // L2 prefetch 4 K-steps ahead (one lane per 64-B tile row).
    if (ks + 4 < KSTEPS && tid < BM) {
      __builtin_prefetch(aG + (size_t)(ks + 4) * 32 + (size_t)tid * DIN, 0, 1);
      __builtin_prefetch(bG + (size_t)(ks + 4) * 32 + (size_t)tid * DIN, 0, 1);
    }

    v16bf bfr[4];
    #pragma unroll
    for (int nt = 0; nt < 4; ++nt)
      bfr[nt] = load_frag(&Bs[buf][waveN * 64 + nt * 16 + lh][kh]);
    #pragma unroll
    for (int mt = 0; mt < 2; ++mt) {
      v16bf a = load_frag(&As[buf][waveM * 32 + mt * 16 + lh][kh]);
      #pragma unroll
      for (int nt = 0; nt < 4; ++nt)
        acc[mt][nt] = __builtin_amdgcn_wmma_f32_16x16x32_bf16(
            false, a, false, bfr[nt], (short)0, acc[mt][nt], false, false);
    }

    asm volatile("s_wait_asynccnt 0x0" ::: "memory");
    __syncthreads();
    buf = nbuf;
  }

  // Epilogue: bias + store fp32 (C/D layout: VGPR i -> M = i + 8*(lane>=16))
  #pragma unroll
  for (int mt = 0; mt < 2; ++mt) {
    int row0 = blockM + waveM * 32 + mt * 16 + ((lane >> 4) << 3);
    #pragma unroll
    for (int nt = 0; nt < 4; ++nt) {
      int col = blockN + waveN * 64 + nt * 16 + lh;
      float bb = bias[col];
      #pragma unroll
      for (int i = 0; i < 8; ++i)
        out[(size_t)(row0 + i) * DOUT + col] = acc[mt][nt][i] + bb;
    }
  }
}

// ---------------------------------------------------------------------------
extern "C" void kernel_launch(void* const* d_in, const int* in_sizes, int n_in,
                              void* d_out, int out_size, void* d_ws,
                              size_t ws_size, hipStream_t stream) {
  const float* x        = (const float*)d_in[0];   // [8,2048,4096]
  const int*   task_id  = (const int*)d_in[1];     // [8]
  const float* base_W   = (const float*)d_in[2];   // [4096,4096]
  const float* base_b   = (const float*)d_in[3];   // [4096]
  const float* Wa       = (const float*)d_in[4];   // [8,8,4096]
  const float* Wb       = (const float*)d_in[5];   // [8,4096,8]
  const float* task_emb = (const float*)d_in[6];   // [9,64]
  const float* gate_W   = (const float*)d_in[7];   // [8,64]
  float* out = (float*)d_out;

  char* ws = (char*)d_ws;
  unsigned short* xb    = (unsigned short*)(ws);
  unsigned short* wb16  = (unsigned short*)(ws + 134217728ull);
  unsigned short* wab   = (unsigned short*)(ws + 167772160ull);
  unsigned short* wball = (unsigned short*)(ws + 168296448ull);
  unsigned short* hg    = (unsigned short*)(ws + 168820736ull);
  float*          gw    = (float*)(ws + 170917888ull);

  // 1) converts (fp32 -> bf16)
  {
    int n4 = (MM * DIN) / 4;
    cvt_f32_bf16_kernel<<<n4 / 256, 256, 0, stream>>>(x, xb, n4);
  }
  {
    int n4 = (DOUT * DIN) / 4;
    cvt_f32_bf16_kernel<<<n4 / 256, 256, 0, stream>>>(base_W, wb16, n4);
  }
  {
    int n4 = (RTOT * DIN) / 4;
    cvt_f32_bf16_kernel<<<n4 / 256, 256, 0, stream>>>(Wa, wab, n4);
  }
  // 2) Wb permute -> [DOUT, 64]
  wb_permute_kernel<<<(NE * DOUT * RE) / 256, 256, 0, stream>>>(Wb, wball);
  // 3) gate softmax
  gate_kernel<<<1, 64, 0, stream>>>(task_id, task_emb, gate_W, gw);
  // 4) hg = SCALING * gw * (x @ Wa^T)   (WMMA, 1024 wave-tiles)
  lora_h_kernel<<<MM / 16 / 4, 128, 0, stream>>>(xb, wab, gw, hg);
  // 5) fused base GEMM + LoRA delta + bias
  dim3 grid(DOUT / BN, MM / BM);
  mmoe_gemm_kernel<<<grid, 256, 0, stream>>>(xb, wb16, hg, wball, base_b, out);
}
